// CausalAttention_69269232549973
// MI455X (gfx1250) — matrix operand order repaired and checked by reading, not compiled, not run on hardware
//
#include <hip/hip_runtime.h>

// Problem constants (match reference)
#define B_    4
#define S_    2048
#define DIN   1024
#define DOUT  1024
#define MROWS (B_ * S_)   // 8192

typedef __attribute__((ext_vector_type(16))) _Float16 v16h;
typedef __attribute__((ext_vector_type(8)))  _Float16 v8h;
typedef __attribute__((ext_vector_type(8)))  float    v8f;

union V16 { v16h v; v8h h[2]; };

// ---------------------------------------------------------------------------
// Kernel 1a: x f32 -> f16
// ---------------------------------------------------------------------------
__global__ void k_cvt_x(const float* __restrict__ x, _Float16* __restrict__ xh, int n) {
    int i = blockIdx.x * blockDim.x + threadIdx.x;
    if (i < n) xh[i] = (_Float16)x[i];
}

// ---------------------------------------------------------------------------
// Kernel 1b: W[k][n] f32 -> wT[which][n][k] f16  (transposed so WMMA B-operand
// reads two runs of 8 contiguous halves along K per lane)
// ---------------------------------------------------------------------------
__global__ void k_cvt_wT(const float* __restrict__ Wq, const float* __restrict__ Wk,
                         const float* __restrict__ Wv, _Float16* __restrict__ wT) {
    int idx   = blockIdx.x * blockDim.x + threadIdx.x;   // over DIN*DOUT
    int which = blockIdx.y;
    const float* W = (which == 0) ? Wq : (which == 1) ? Wk : Wv;
    int k = idx / DOUT;
    int n = idx % DOUT;
    wT[(size_t)which * DIN * DOUT + (size_t)n * DIN + k] = (_Float16)W[(size_t)k * DOUT + n];
}

// ---------------------------------------------------------------------------
// Kernel 2: QKV projection GEMM, 32x64 output tile per wave (2x4 register
// blocking: 8 WMMAs per k-step off 12 b128 loads).
// Q,K -> row-major f16. V -> transposed f16 Vt[b][d][s].
// ---------------------------------------------------------------------------
__global__ void __launch_bounds__(32) k_qkv(const _Float16* __restrict__ xh,
                                            const _Float16* __restrict__ wT,
                                            _Float16* __restrict__ Qh,
                                            _Float16* __restrict__ Kh,
                                            _Float16* __restrict__ Vt) {
    const int lane = threadIdx.x;
    const int r  = lane & 15;
    const int hh = lane >> 4;
    const int m0 = blockIdx.x * 32;
    const int n0 = blockIdx.y * 64;
    const int which = blockIdx.z;

    const _Float16* a0 = xh + (size_t)(m0 + r) * DIN;                 // rows m0..m0+15
    const _Float16* a1 = a0 + (size_t)16 * DIN;                       // rows m0+16..
    const _Float16* b0 = wT + (size_t)which * DIN * DOUT + (size_t)(n0 + r) * DIN;

    v8f acc[2][4] = {};
    for (int k0 = 0; k0 < DIN; k0 += 32) {
        V16 a[2], b[4];
        a[0].h[0] = *(const v8h*)(a0 + k0 + hh * 8);
        a[0].h[1] = *(const v8h*)(a0 + k0 + 16 + hh * 8);
        a[1].h[0] = *(const v8h*)(a1 + k0 + hh * 8);
        a[1].h[1] = *(const v8h*)(a1 + k0 + 16 + hh * 8);
#pragma unroll
        for (int ni = 0; ni < 4; ++ni)
        {
            b[ni].h[0] = *(const v8h*)(b0 + (size_t)ni * 16 * DIN + k0 + hh * 8);
            b[ni].h[1] = *(const v8h*)(b0 + (size_t)ni * 16 * DIN + k0 + 16 + hh * 8);
        }
#pragma unroll
        for (int mi = 0; mi < 2; ++mi)
#pragma unroll
            for (int ni = 0; ni < 4; ++ni)
                acc[mi][ni] = __builtin_amdgcn_wmma_f32_16x16x32_f16(
                    false, a[mi].v, false, b[ni].v, (short)0, acc[mi][ni], false, false);
    }

    if (which < 2) {
        _Float16* dst = (which == 0) ? Qh : Kh;
#pragma unroll
        for (int mi = 0; mi < 2; ++mi)
#pragma unroll
            for (int ni = 0; ni < 4; ++ni)
#pragma unroll
                for (int v = 0; v < 8; ++v)
                    dst[(size_t)(m0 + mi * 16 + v + 8 * hh) * DOUT + (n0 + ni * 16 + r)] =
                        (_Float16)acc[mi][ni][v];
    } else {
        const int b  = m0 / S_;
        const int s0 = m0 % S_;
#pragma unroll
        for (int mi = 0; mi < 2; ++mi)
#pragma unroll
            for (int ni = 0; ni < 4; ++ni)
#pragma unroll
                for (int v = 0; v < 8; ++v)
                    Vt[(size_t)b * DOUT * S_ + (size_t)(n0 + ni * 16 + r) * S_ +
                       (s0 + mi * 16 + v + 8 * hh)] = (_Float16)acc[mi][ni][v];
    }
}

// ---------------------------------------------------------------------------
// Kernel 3: scores = Q . K^T, 32 queries x 64 keys per wave, causal skip at
// block granularity. Above-diagonal lanes inside kept blocks write garbage
// that the softmax never reads.
// ---------------------------------------------------------------------------
__global__ void __launch_bounds__(32) k_scores(const _Float16* __restrict__ Qh,
                                               const _Float16* __restrict__ Kh,
                                               float* __restrict__ sc) {
    const int j0 = blockIdx.x * 64;   // key block
    const int i0 = blockIdx.y * 32;   // query block
    const int b  = blockIdx.z;
    if (j0 > i0 + 31) return;         // block fully above the diagonal

    const int lane = threadIdx.x;
    const int r  = lane & 15;
    const int hh = lane >> 4;

    const _Float16* a0 = Qh + (size_t)(b * S_ + i0 + r) * DOUT;
    const _Float16* a1 = a0 + (size_t)16 * DOUT;
    const _Float16* b0 = Kh + (size_t)(b * S_ + j0 + r) * DOUT;

    v8f acc[2][4] = {};
    for (int k0 = 0; k0 < DOUT; k0 += 32) {
        V16 a[2], bm[4];
        a[0].h[0] = *(const v8h*)(a0 + k0 + hh * 8);
        a[0].h[1] = *(const v8h*)(a0 + k0 + 16 + hh * 8);
        a[1].h[0] = *(const v8h*)(a1 + k0 + hh * 8);
        a[1].h[1] = *(const v8h*)(a1 + k0 + 16 + hh * 8);
#pragma unroll
        for (int ni = 0; ni < 4; ++ni)
        {
            bm[ni].h[0] = *(const v8h*)(b0 + (size_t)ni * 16 * DOUT + k0 + hh * 8);
            bm[ni].h[1] = *(const v8h*)(b0 + (size_t)ni * 16 * DOUT + k0 + 16 + hh * 8);
        }
#pragma unroll
        for (int mi = 0; mi < 2; ++mi)
#pragma unroll
            for (int ni = 0; ni < 4; ++ni)
                acc[mi][ni] = __builtin_amdgcn_wmma_f32_16x16x32_f16(
                    false, a[mi].v, false, bm[ni].v, (short)0, acc[mi][ni], false, false);
    }

    float* out = sc + (size_t)b * S_ * S_;
#pragma unroll
    for (int mi = 0; mi < 2; ++mi)
#pragma unroll
        for (int ni = 0; ni < 4; ++ni)
#pragma unroll
            for (int v = 0; v < 8; ++v)
                out[(size_t)(i0 + mi * 16 + v + 8 * hh) * S_ + (j0 + ni * 16 + r)] =
                    acc[mi][ni][v];
}

// ---------------------------------------------------------------------------
// Kernel 4: causal softmax per row (mask -> scale by 1/sqrt(d) -> softmax),
// writes f16 probabilities, zero-padded up to the 32-key WMMA step boundary.
// ---------------------------------------------------------------------------
__global__ void __launch_bounds__(256) k_softmax(const float* __restrict__ sc,
                                                 _Float16* __restrict__ Ph) {
    __shared__ float red[256];
    const int row = blockIdx.x;           // 0 .. MROWS-1
    const int b = row / S_;
    const int i = row % S_;
    const float* srow = sc + (size_t)b * S_ * S_ + (size_t)i * S_;
    _Float16*    prow = Ph + (size_t)b * S_ * S_ + (size_t)i * S_;
    const int tid = threadIdx.x;
    const float scale = 0.03125f;         // 1/sqrt(1024)

    float lm = -3.4e38f;
    for (int j = tid; j <= i; j += 256) lm = fmaxf(lm, srow[j] * scale);
    red[tid] = lm; __syncthreads();
    for (int o = 128; o > 0; o >>= 1) {
        if (tid < o) red[tid] = fmaxf(red[tid], red[tid + o]);
        __syncthreads();
    }
    const float m = red[0]; __syncthreads();

    float ls = 0.f;
    for (int j = tid; j <= i; j += 256) ls += __expf(srow[j] * scale - m);
    red[tid] = ls; __syncthreads();
    for (int o = 128; o > 0; o >>= 1) {
        if (tid < o) red[tid] += red[tid + o];
        __syncthreads();
    }
    const float inv = 1.0f / red[0];

    int lpad = ((i >> 4) + 1) << 4;       // pad row to its tile boundary,
    lpad = (lpad + 31) & ~31;             // rounded up to the 32-key WMMA step
    for (int j = tid; j < lpad; j += 256) {
        float p = (j <= i) ? __expf(srow[j] * scale - m) * inv : 0.f;
        prow[j] = (_Float16)p;
    }
}

// ---------------------------------------------------------------------------
// Kernel 5: out = P . V, 32 queries x 64 out-cols per wave. Key loop runs to
// i0+32 (the softmax zero-pads P exactly that far for these rows).
// ---------------------------------------------------------------------------
__global__ void __launch_bounds__(32) k_pv(const _Float16* __restrict__ Ph,
                                           const _Float16* __restrict__ Vt,
                                           float* __restrict__ out) {
    const int n0 = blockIdx.x * 64;
    const int i0 = blockIdx.y * 32;
    const int b  = blockIdx.z;
    const int lane = threadIdx.x;
    const int r  = lane & 15;
    const int hh = lane >> 4;

    const _Float16* a0 = Ph + (size_t)b * S_ * S_ + (size_t)(i0 + r) * S_;
    const _Float16* a1 = a0 + (size_t)16 * S_;
    const _Float16* b0 = Vt + (size_t)b * DOUT * S_ + (size_t)(n0 + r) * S_;
    const int kend = i0 + 32;             // causal boundary of this query block

    v8f acc[2][4] = {};
    for (int k0 = 0; k0 < kend; k0 += 32) {
        V16 a[2], bm[4];
        a[0].h[0] = *(const v8h*)(a0 + k0 + hh * 8);
        a[0].h[1] = *(const v8h*)(a0 + k0 + 16 + hh * 8);
        a[1].h[0] = *(const v8h*)(a1 + k0 + hh * 8);
        a[1].h[1] = *(const v8h*)(a1 + k0 + 16 + hh * 8);
#pragma unroll
        for (int ni = 0; ni < 4; ++ni)
        {
            bm[ni].h[0] = *(const v8h*)(b0 + (size_t)ni * 16 * S_ + k0 + hh * 8);
            bm[ni].h[1] = *(const v8h*)(b0 + (size_t)ni * 16 * S_ + k0 + 16 + hh * 8);
        }
#pragma unroll
        for (int mi = 0; mi < 2; ++mi)
#pragma unroll
            for (int ni = 0; ni < 4; ++ni)
                acc[mi][ni] = __builtin_amdgcn_wmma_f32_16x16x32_f16(
                    false, a[mi].v, false, bm[ni].v, (short)0, acc[mi][ni], false, false);
    }

    float* obase = out + (size_t)(b * S_ + i0) * DOUT;
#pragma unroll
    for (int mi = 0; mi < 2; ++mi)
#pragma unroll
        for (int ni = 0; ni < 4; ++ni)
#pragma unroll
            for (int v = 0; v < 8; ++v)
                obase[(size_t)(mi * 16 + v + 8 * hh) * DOUT + (n0 + ni * 16 + r)] =
                    acc[mi][ni][v];
}

// ---------------------------------------------------------------------------
// Host-side launch
// ---------------------------------------------------------------------------
extern "C" void kernel_launch(void* const* d_in, const int* in_sizes, int n_in,
                              void* d_out, int out_size, void* d_ws, size_t ws_size,
                              hipStream_t stream) {
    const float* x  = (const float*)d_in[0];
    const float* Wq = (const float*)d_in[1];
    const float* Wk = (const float*)d_in[2];
    const float* Wv = (const float*)d_in[3];
    float* out = (float*)d_out;

    // workspace layout (bytes)
    const size_t SZ_XH = (size_t)MROWS * DIN * 2;        // 16 MB
    const size_t SZ_WT = (size_t)3 * DIN * DOUT * 2;     //  6 MB
    const size_t SZ_QH = (size_t)MROWS * DOUT * 2;       // 16 MB
    const size_t SZ_KH = SZ_QH;                          // 16 MB
    const size_t SZ_VT = SZ_QH;                          // 16 MB
    const size_t SZ_SC = (size_t)B_ * S_ * S_ * 4;       // 64 MB

    char* ws = (char*)d_ws;
    _Float16* xh = (_Float16*)(ws);
    _Float16* wT = (_Float16*)(ws + SZ_XH);
    _Float16* Qh = (_Float16*)(ws + SZ_XH + SZ_WT);
    _Float16* Kh = (_Float16*)(ws + SZ_XH + SZ_WT + SZ_QH);
    _Float16* Vt = (_Float16*)(ws + SZ_XH + SZ_WT + SZ_QH + SZ_KH);
    float*    sc = (float*)   (ws + SZ_XH + SZ_WT + SZ_QH + SZ_KH + SZ_VT);
    _Float16* Ph = (_Float16*)(ws + SZ_XH + SZ_WT + SZ_QH + SZ_KH + SZ_VT + SZ_SC);

    // 1) convert inputs to f16 (x row-major, W transposed)
    {
        int n = MROWS * DIN;
        k_cvt_x<<<dim3((n + 255) / 256), dim3(256), 0, stream>>>(x, xh, n);
        k_cvt_wT<<<dim3((DIN * DOUT) / 256, 3), dim3(256), 0, stream>>>(Wq, Wk, Wv, wT);
    }
    // 2) QKV projections (WMMA, 32x64 tiles)
    k_qkv<<<dim3(MROWS / 32, DOUT / 64, 3), dim3(32), 0, stream>>>(xh, wT, Qh, Kh, Vt);
    // 3) causal score tiles (WMMA, 32x64 tiles)
    k_scores<<<dim3(S_ / 64, S_ / 32, B_), dim3(32), 0, stream>>>(Qh, Kh, sc);
    // 4) masked, scaled softmax -> f16 P
    k_softmax<<<dim3(MROWS), dim3(256), 0, stream>>>(sc, Ph);
    // 5) P . V (WMMA, 32x64 tiles) -> fp32 output
    k_pv<<<dim3(DOUT / 64, S_ / 32, B_), dim3(32), 0, stream>>>(Ph, Vt, out);
}
